// LSTMDecoderWithAttention_53085795779128
// MI455X (gfx1250) — matrix operand order repaired and checked by reading, not compiled, not run on hardware
//
#include <hip/hip_runtime.h>
#include <hip/hip_bf16.h>
#include <stdint.h>

// ---------------- problem constants ----------------
#define Bc   64
#define Tc   32
#define Hc   1024
#define Ec   512
#define Vc   32000
#define Sc   49
#define QLc  32
#define NHc  8
#define DHc  128
#define SCALEc 0.08838834764831845f   // DH^-0.5

// ---------------- WMMA types ----------------
typedef __attribute__((ext_vector_type(16))) __bf16 v16bf;
typedef __attribute__((ext_vector_type(8)))  float  v8f;

union Frag16 { v16bf v; unsigned u[8]; uint4 q[2]; };

__device__ __forceinline__ unsigned short f2bf(float x) {
    unsigned u = __float_as_uint(x);
    u += 0x7FFFu + ((u >> 16) & 1u);                       // RNE
    return (unsigned short)(u >> 16);
}
__device__ __forceinline__ unsigned packbf2(float lo, float hi) {
    return (unsigned)f2bf(lo) | ((unsigned)f2bf(hi) << 16);
}

// ============================================================
// WMMA GEMM:  C(M,N) = A(M,K)bf16 @ W(N,K)bf16^T [+bias] [+=C]
// One wave -> 32x32 C tile (2x2 WMMA tiles, 4 accumulators).
// Block = 256 thr = 8 waves = 64 rows x 128 cols.
// grid = (M/64, N/128).  M%64==0, N%128==0, K%32==0 at all call sites.
// Inner loop per 32-K: 8x global_load_b128 + 4x v_wmma (A,B each reused 2x).
// ============================================================
__global__ __launch_bounds__(256)
void gemm_bf16_wmma(const unsigned short* __restrict__ A, int lda,
                    const unsigned short* __restrict__ W, int ldw,
                    float* __restrict__ C, long ldc,
                    unsigned short* __restrict__ Cbf,     // non-null -> bf16 output
                    const float* __restrict__ bias,
                    int K, int accumulate)
{
    const int lane = threadIdx.x & 31;
    const int wave = threadIdx.x >> 5;
    const int half = lane >> 4;                  // 0/1
    const int l15  = lane & 15;
    const int tile_m = blockIdx.x * 64  + (wave & 1) * 32;
    const int tile_n = blockIdx.y * 128 + (wave >> 1) * 32;

    // A rows for the two M sub-tiles; 16-bit A 16x32 layout:
    // lane half=0 holds K{0..7,16..23}, half=1 holds K{8..15,24..31}.
    const unsigned short* Ar0 = A + (size_t)(tile_m + l15)      * lda + half * 8;
    const unsigned short* Ar1 = A + (size_t)(tile_m + 16 + l15) * lda + half * 8;
    // B cols for the two N sub-tiles; same K layout as A: half=0 K{0..7,16..23}, half=1 K{8..15,24..31}.
    const unsigned short* Wr0 = W + (size_t)(tile_n + l15)      * ldw + half * 8;
    const unsigned short* Wr1 = W + (size_t)(tile_n + 16 + l15) * ldw + half * 8;

    v8f acc00, acc01, acc10, acc11;
    {
        const float b0 = bias ? bias[tile_n + l15]      : 0.0f;
        const float b1 = bias ? bias[tile_n + 16 + l15] : 0.0f;
        const int m0 = tile_m + half * 8;
        const int m1 = tile_m + 16 + half * 8;
        const int n0 = tile_n + l15;
        const int n1 = tile_n + 16 + l15;
#pragma unroll
        for (int r = 0; r < 8; ++r) {
            float c00 = 0.f, c01 = 0.f, c10 = 0.f, c11 = 0.f;
            if (accumulate) {
                c00 = C[(size_t)(m0 + r) * ldc + n0];
                c01 = C[(size_t)(m0 + r) * ldc + n1];
                c10 = C[(size_t)(m1 + r) * ldc + n0];
                c11 = C[(size_t)(m1 + r) * ldc + n1];
            }
            acc00[r] = c00 + b0;  acc01[r] = c01 + b1;
            acc10[r] = c10 + b0;  acc11[r] = c11 + b1;
        }
    }

    for (int k0 = 0; k0 < K; k0 += 32) {
        Frag16 a0, a1, b0, b1;
        a0.q[0] = *(const uint4*)(Ar0 + k0);
        a0.q[1] = *(const uint4*)(Ar0 + k0 + 16);
        a1.q[0] = *(const uint4*)(Ar1 + k0);
        a1.q[1] = *(const uint4*)(Ar1 + k0 + 16);
        b0.q[0] = *(const uint4*)(Wr0 + k0);
        b0.q[1] = *(const uint4*)(Wr0 + k0 + 16);
        b1.q[0] = *(const uint4*)(Wr1 + k0);
        b1.q[1] = *(const uint4*)(Wr1 + k0 + 16);

        acc00 = __builtin_amdgcn_wmma_f32_16x16x32_bf16(false, a0.v, false, b0.v, (short)0, acc00, false, false);
        acc01 = __builtin_amdgcn_wmma_f32_16x16x32_bf16(false, a0.v, false, b1.v, (short)0, acc01, false, false);
        acc10 = __builtin_amdgcn_wmma_f32_16x16x32_bf16(false, a1.v, false, b0.v, (short)0, acc10, false, false);
        acc11 = __builtin_amdgcn_wmma_f32_16x16x32_bf16(false, a1.v, false, b1.v, (short)0, acc11, false, false);
    }

    // store (f32 C/D layout: VGPR r -> M = tile_m + half*8 + r, N = lane&15)
    const int m0 = tile_m + half * 8;
    const int m1 = tile_m + 16 + half * 8;
    const int n0 = tile_n + l15;
    const int n1 = tile_n + 16 + l15;
    if (Cbf) {
#pragma unroll
        for (int r = 0; r < 8; ++r) {
            Cbf[(size_t)(m0 + r) * ldc + n0] = f2bf(acc00[r]);
            Cbf[(size_t)(m0 + r) * ldc + n1] = f2bf(acc01[r]);
            Cbf[(size_t)(m1 + r) * ldc + n0] = f2bf(acc10[r]);
            Cbf[(size_t)(m1 + r) * ldc + n1] = f2bf(acc11[r]);
        }
    } else {
#pragma unroll
        for (int r = 0; r < 8; ++r) {
            C[(size_t)(m0 + r) * ldc + n0] = acc00[r];
            C[(size_t)(m0 + r) * ldc + n1] = acc01[r];
            C[(size_t)(m1 + r) * ldc + n0] = acc10[r];
            C[(size_t)(m1 + r) * ldc + n1] = acc11[r];
        }
    }
}

// ============================================================
// f32 -> bf16 conversion (weights + static activations, once per launch)
// ============================================================
__global__ __launch_bounds__(256)
void cvt_bf16_kernel(const float* __restrict__ src, unsigned short* __restrict__ dst, int n)
{
    int i = blockIdx.x * 256 + threadIdx.x;
    if (i < n) dst[i] = f2bf(src[i]);
}

// ============================================================
// State init: h(bf16)=h0, c=c0, coverage=0, cov_loss=0
// ============================================================
__global__ __launch_bounds__(256)
void init_state_kernel(const float* __restrict__ h0, const float* __restrict__ c0,
                       unsigned short* __restrict__ hbf, float* __restrict__ c,
                       float* __restrict__ coverage, float* __restrict__ covloss)
{
    int i = blockIdx.x * 256 + threadIdx.x;          // B*H threads
    hbf[i] = f2bf(h0[i]);
    c[i]   = c0[i];
    if (i < Bc * Sc) coverage[i] = 0.0f;
    if (i == 0) covloss[0] = 0.0f;
}

// ============================================================
// Per-head attention: one wave per (b, head).
// q:(B,H)f32  kmem/vmem:(B,Smem,H)f32 laid (b,s,head,dh)
// ctx:(B,H) bf16   alpha_out:(B,NH,Smem) optional
// ============================================================
__global__ __launch_bounds__(256)
void attn_kernel(const float* __restrict__ q,
                 const float* __restrict__ kmem,
                 const float* __restrict__ vmem,
                 const float* __restrict__ coverage,    // (B,Smem) or null
                 const float* __restrict__ cov_scale,   // scalar or null
                 unsigned short* __restrict__ ctx,      // bf16 out
                 float* __restrict__ alpha_out,         // may be null
                 int Smem)
{
    __shared__ float sc[8][64];
    const int lane = threadIdx.x & 31;
    const int wslt = threadIdx.x >> 5;
    const int wg   = blockIdx.x * 8 + wslt;     // 0 .. B*NH-1
    const int b    = wg / NHc;
    const int hh   = wg % NHc;

    const float4 qv = *(const float4*)(q + (size_t)b * Hc + hh * DHc + lane * 4);
    const float cs  = cov_scale ? cov_scale[0] : 0.0f;

    // ---- scores ----
    for (int s = 0; s < Smem; ++s) {
        const float4 kv = *(const float4*)(kmem + ((size_t)b * Smem + s) * Hc + hh * DHc + lane * 4);
        float d = qv.x * kv.x + qv.y * kv.y + qv.z * kv.z + qv.w * kv.w;
#pragma unroll
        for (int m = 16; m > 0; m >>= 1) d += __shfl_xor(d, m, 32);
        float sco = d * SCALEc;
        if (coverage) sco += cs * coverage[(size_t)b * Smem + s];
        if (lane == 0) sc[wslt][s] = sco;
    }
    __syncthreads();

    // ---- softmax over S ----
    float lmax = -1e30f;
    for (int s = lane; s < Smem; s += 32) lmax = fmaxf(lmax, sc[wslt][s]);
#pragma unroll
    for (int m = 16; m > 0; m >>= 1) lmax = fmaxf(lmax, __shfl_xor(lmax, m, 32));
    float lsum = 0.0f;
    for (int s = lane; s < Smem; s += 32) {
        float e = __expf(sc[wslt][s] - lmax);
        sc[wslt][s] = e;
        lsum += e;
    }
#pragma unroll
    for (int m = 16; m > 0; m >>= 1) lsum += __shfl_xor(lsum, m, 32);
    const float inv = 1.0f / lsum;
    for (int s = lane; s < Smem; s += 32) {
        float a = sc[wslt][s] * inv;
        sc[wslt][s] = a;
        if (alpha_out) alpha_out[((size_t)b * NHc + hh) * Smem + s] = a;
    }
    __syncthreads();

    // ---- ctx = alpha @ V, emitted as bf16 (GEMM A operand) ----
    float4 accv = make_float4(0.f, 0.f, 0.f, 0.f);
    for (int s = 0; s < Smem; ++s) {
        const float a = sc[wslt][s];
        const float4 vv = *(const float4*)(vmem + ((size_t)b * Smem + s) * Hc + hh * DHc + lane * 4);
        accv.x += a * vv.x; accv.y += a * vv.y; accv.z += a * vv.z; accv.w += a * vv.w;
    }
    uint2 packed = make_uint2(packbf2(accv.x, accv.y), packbf2(accv.z, accv.w));
    *(uint2*)(ctx + (size_t)b * Hc + hh * DHc + lane * 4) = packed;
}

// ============================================================
// Coverage loss + update (single block, deterministic).
// cov_accum += sum_{b,s} min(mean_h alpha, coverage) / (B*T)
// ============================================================
__global__ __launch_bounds__(256)
void coverage_kernel(const float* __restrict__ alpha,
                     float* __restrict__ coverage,
                     float* __restrict__ cov_accum)
{
    __shared__ float red[256];
    const int tid = threadIdx.x;
    float part = 0.0f;
    for (int i = tid; i < Bc * Sc; i += 256) {
        const int b = i / Sc, s = i % Sc;
        float am = 0.0f;
#pragma unroll
        for (int h = 0; h < NHc; ++h) am += alpha[((size_t)b * NHc + h) * Sc + s];
        am *= (1.0f / NHc);
        const float cov = coverage[i];
        part += fminf(am, cov);
        coverage[i] = cov + am;
    }
    red[tid] = part;
    __syncthreads();
    for (int off = 128; off > 0; off >>= 1) {
        if (tid < off) red[tid] += red[tid + off];
        __syncthreads();
    }
    if (tid == 0) cov_accum[0] += red[0] * (1.0f / (Bc * Tc));
}

// ============================================================
// x = [embed(token_t), img_ctx, q_ctx] -> (B, 2560) bf16 (GEMM A operand)
// ============================================================
__global__ __launch_bounds__(256)
void build_x_kernel(const float* __restrict__ emb, const int* __restrict__ target_seq, int t,
                    const float* __restrict__ ctx_i, const float* __restrict__ ctx_q,
                    unsigned short* __restrict__ x)
{
    int i = blockIdx.x * 256 + threadIdx.x;          // B * 2560
    const int b = i / 2560, j = i % 2560;
    float v;
    if (j < Ec) {
        const int tok = target_seq[b * Tc + t];
        v = emb[(size_t)tok * Ec + j];
    } else if (j < Ec + Hc) {
        v = ctx_i[(size_t)b * Hc + (j - Ec)];
    } else {
        v = ctx_q[(size_t)b * Hc + (j - Ec - Hc)];
    }
    x[i] = f2bf(v);
}

// ============================================================
// LSTM cell pointwise (torch gate order i,f,g,o); c f32, h bf16.
// ============================================================
__global__ __launch_bounds__(256)
void lstm_cell_kernel(const float* __restrict__ gates,
                      float* __restrict__ c, unsigned short* __restrict__ hbf)
{
    const int i = blockIdx.x * 256 + threadIdx.x;    // B*H threads
    const int b = i >> 10;
    const int j = i & 1023;
    const float* g = gates + (size_t)b * (4 * Hc);
    const float gi = g[j], gf = g[Hc + j], gg = g[2 * Hc + j], go = g[3 * Hc + j];
    const float si = 1.0f / (1.0f + __expf(-gi));
    const float sf = 1.0f / (1.0f + __expf(-gf));
    const float so = 1.0f / (1.0f + __expf(-go));
    const float cc = sf * c[i] + si * tanhf(gg);
    c[i]   = cc;
    hbf[i] = f2bf(so * tanhf(cc));
}

// ============================================================
// host-side launcher
// ============================================================
static inline void run_gemm(hipStream_t s,
                            const unsigned short* A, int lda,
                            const unsigned short* W, int ldw,
                            float* C, long ldc, unsigned short* Cbf,
                            const float* bias, int M, int N, int K, int acc)
{
    dim3 grid(M / 64, N / 128);
    gemm_bf16_wmma<<<grid, 256, 0, s>>>(A, lda, W, ldw, C, ldc, Cbf, bias, K, acc);
}

extern "C" void kernel_launch(void* const* d_in, const int* in_sizes, int n_in,
                              void* d_out, int out_size, void* d_ws, size_t ws_size,
                              hipStream_t stream)
{
    (void)in_sizes; (void)n_in; (void)out_size; (void)ws_size;

    const float* h0    = (const float*)d_in[0];
    const float* c0    = (const float*)d_in[1];
    const float* img   = (const float*)d_in[2];    // (B,S,H)
    const float* qh    = (const float*)d_in[3];    // (B,QL,H)
    const int*   tseq  = (const int*)  d_in[4];    // (B,T)
    const float* emb   = (const float*)d_in[5];    // (V,E)
    const float* Wq_i  = (const float*)d_in[6];
    const float* Wk_i  = (const float*)d_in[7];
    const float* Wv_i  = (const float*)d_in[8];
    const float* Wo_i  = (const float*)d_in[9];
    const float* covsc = (const float*)d_in[10];
    const float* Wq_q  = (const float*)d_in[11];
    const float* Wk_q  = (const float*)d_in[12];
    const float* Wv_q  = (const float*)d_in[13];
    const float* Wo_q  = (const float*)d_in[14];
    const float* W_ih  = (const float*)d_in[15];   // (4H, E+2H)
    const float* W_hh  = (const float*)d_in[16];   // (4H, H)
    const float* b_ih  = (const float*)d_in[17];
    const float* b_hh  = (const float*)d_in[18];
    const float* Wout  = (const float*)d_in[19];   // (E, H)
    const float* bout  = (const float*)d_in[20];

    float* logits  = (float*)d_out;                          // (B,T,V)
    float* covloss = (float*)d_out + (size_t)Bc * Tc * Vc;   // scalar

    // ---------- carve workspace ----------
    char* p = (char*)d_ws;
    auto carve = [&](size_t bytes) -> void* {
        void* r = (void*)p;
        p += (bytes + 255) & ~(size_t)255;
        return r;
    };
    const size_t HH = (size_t)Hc * Hc;
    unsigned short* wq_i_b = (unsigned short*)carve(HH * 2);
    unsigned short* wk_i_b = (unsigned short*)carve(HH * 2);
    unsigned short* wv_i_b = (unsigned short*)carve(HH * 2);
    unsigned short* wo_i_b = (unsigned short*)carve(HH * 2);
    unsigned short* wq_q_b = (unsigned short*)carve(HH * 2);
    unsigned short* wk_q_b = (unsigned short*)carve(HH * 2);
    unsigned short* wv_q_b = (unsigned short*)carve(HH * 2);
    unsigned short* wo_q_b = (unsigned short*)carve(HH * 2);
    unsigned short* wih_b  = (unsigned short*)carve((size_t)4 * Hc * 2560 * 2);
    unsigned short* whh_b  = (unsigned short*)carve((size_t)4 * Hc * Hc * 2);
    unsigned short* wout_b = (unsigned short*)carve((size_t)Ec * Hc * 2);
    unsigned short* emb_b  = (unsigned short*)carve((size_t)Vc * Ec * 2);
    unsigned short* img_b  = (unsigned short*)carve((size_t)Bc * Sc  * Hc * 2);
    unsigned short* qh_b   = (unsigned short*)carve((size_t)Bc * QLc * Hc * 2);
    float* k_img = (float*)carve((size_t)Bc * Sc  * Hc * 4);
    float* v_img = (float*)carve((size_t)Bc * Sc  * Hc * 4);
    float* k_q   = (float*)carve((size_t)Bc * QLc * Hc * 4);
    float* v_q   = (float*)carve((size_t)Bc * QLc * Hc * 4);
    unsigned short* hbf   = (unsigned short*)carve((size_t)Bc * Hc * 2);
    float*          cbuf  = (float*)carve((size_t)Bc * Hc * 4);
    float*          qi    = (float*)carve((size_t)Bc * Hc * 4);
    float*          qq    = (float*)carve((size_t)Bc * Hc * 4);
    unsigned short* ctx_i = (unsigned short*)carve((size_t)Bc * Hc * 2);
    unsigned short* ctx_q = (unsigned short*)carve((size_t)Bc * Hc * 2);
    float*          prj_i = (float*)carve((size_t)Bc * Hc * 4);
    float*          prj_q = (float*)carve((size_t)Bc * Hc * 4);
    float*          alpha = (float*)carve((size_t)Bc * NHc * Sc * 4);
    float*          cover = (float*)carve((size_t)Bc * Sc * 4);
    unsigned short* xbuf  = (unsigned short*)carve((size_t)Bc * 2560 * 2);
    float*          gates = (float*)carve((size_t)Bc * 4 * Hc * 4);
    unsigned short* ebuf  = (unsigned short*)carve((size_t)Bc * Ec * 2);

    // ---------- conversions (once per launch) ----------
    auto cvt = [&](const float* src, unsigned short* dst, int n) {
        cvt_bf16_kernel<<<(n + 255) / 256, 256, 0, stream>>>(src, dst, n);
    };
    cvt(Wq_i, wq_i_b, (int)HH);  cvt(Wk_i, wk_i_b, (int)HH);
    cvt(Wv_i, wv_i_b, (int)HH);  cvt(Wo_i, wo_i_b, (int)HH);
    cvt(Wq_q, wq_q_b, (int)HH);  cvt(Wk_q, wk_q_b, (int)HH);
    cvt(Wv_q, wv_q_b, (int)HH);  cvt(Wo_q, wo_q_b, (int)HH);
    cvt(W_ih, wih_b, 4 * Hc * 2560);
    cvt(W_hh, whh_b, 4 * Hc * Hc);
    cvt(Wout, wout_b, Ec * Hc);
    cvt(emb,  emb_b,  Vc * Ec);
    cvt(img,  img_b,  Bc * Sc  * Hc);
    cvt(qh,   qh_b,   Bc * QLc * Hc);

    // ---------- init state ----------
    init_state_kernel<<<(Bc * Hc) / 256, 256, 0, stream>>>(h0, c0, hbf, cbuf, cover, covloss);

    // ---------- loop-invariant K/V projections (f32 out for VALU attention) ----------
    run_gemm(stream, img_b, Hc, wk_i_b, Hc, k_img, Hc, nullptr, nullptr, Bc * Sc,  Hc, Hc, 0);
    run_gemm(stream, img_b, Hc, wv_i_b, Hc, v_img, Hc, nullptr, nullptr, Bc * Sc,  Hc, Hc, 0);
    run_gemm(stream, qh_b,  Hc, wk_q_b, Hc, k_q,   Hc, nullptr, nullptr, Bc * QLc, Hc, Hc, 0);
    run_gemm(stream, qh_b,  Hc, wv_q_b, Hc, v_q,   Hc, nullptr, nullptr, Bc * QLc, Hc, Hc, 0);

    // ---------- decode loop ----------
    for (int t = 0; t < Tc; ++t) {
        // q projections (f32 out -> attention VALU kernel)
        run_gemm(stream, hbf, Hc, wq_i_b, Hc, qi, Hc, nullptr, nullptr, Bc, Hc, Hc, 0);
        run_gemm(stream, hbf, Hc, wq_q_b, Hc, qq, Hc, nullptr, nullptr, Bc, Hc, Hc, 0);

        // attentions (img side uses coverage bias + records alpha); ctx out bf16
        attn_kernel<<<(Bc * NHc) / 8, 256, 0, stream>>>(qi, k_img, v_img, cover, covsc,
                                                        ctx_i, alpha, Sc);
        attn_kernel<<<(Bc * NHc) / 8, 256, 0, stream>>>(qq, k_q, v_q, nullptr, nullptr,
                                                        ctx_q, nullptr, QLc);
        // coverage loss (uses old coverage) + coverage update
        coverage_kernel<<<1, 256, 0, stream>>>(alpha, cover, covloss);

        // output projections of attention (f32 out -> build_x)
        run_gemm(stream, ctx_i, Hc, wo_i_b, Hc, prj_i, Hc, nullptr, nullptr, Bc, Hc, Hc, 0);
        run_gemm(stream, ctx_q, Hc, wo_q_b, Hc, prj_q, Hc, nullptr, nullptr, Bc, Hc, Hc, 0);

        // x = [embed_t, img_ctx, q_ctx]  (bf16 out)
        build_x_kernel<<<(Bc * 2560) / 256, 256, 0, stream>>>(emb, tseq, t, prj_i, prj_q, xbuf);

        // gates = x @ W_ih^T + b_ih + h @ W_hh^T + b_hh   (f32)
        run_gemm(stream, xbuf, 2560, wih_b, 2560, gates, 4 * Hc, nullptr, b_ih, Bc, 4 * Hc, 2560, 0);
        run_gemm(stream, hbf,  Hc,   whh_b, Hc,   gates, 4 * Hc, nullptr, b_hh, Bc, 4 * Hc, Hc,   1);

        // pointwise cell -> c (f32), h (bf16) in place
        lstm_cell_kernel<<<(Bc * Hc) / 256, 256, 0, stream>>>(gates, cbuf, hbf);

        // e = h @ Wout^T + bout  (bf16 out) ; logits_t = e @ emb^T  (tied weights)
        run_gemm(stream, hbf, Hc, wout_b, Hc, nullptr, Ec, ebuf, bout, Bc, Ec, Hc, 0);
        run_gemm(stream, ebuf, Ec, emb_b, Ec,
                 logits + (size_t)t * Vc, (long)Tc * Vc, nullptr, nullptr, Bc, Vc, Ec, 0);
    }
}
